// ProbAttention_30107720745426
// MI455X (gfx1250) — matrix operand order repaired and checked by reading, not compiled, not run on hardware
//
#include <hip/hip_runtime.h>

typedef _Float16 v16h __attribute__((ext_vector_type(16)));
typedef _Float16 v8h  __attribute__((ext_vector_type(8)));
typedef _Float16 h4   __attribute__((ext_vector_type(4)));
typedef _Float16 h2   __attribute__((ext_vector_type(2)));
typedef float    v8f  __attribute__((ext_vector_type(8)));

namespace {
constexpr int kB = 4, kL = 2048, kH = 8, kD = 64;
constexpr int BC = 64;          // keys per inner iteration
constexpr int BR = 64;          // query rows per workgroup (4 waves * 16)
constexpr int KP = kD + 8;      // padded Ksh row (halves) -> 144B stride, conflict-free
constexpr int VP = BC + 8;      // padded Vt row  (halves) -> 144B stride
constexpr int PP = BC + 8;      // padded Psh row (halves) -> 144B stride
}

static __device__ __forceinline__ v16h cat8(const v8h lo, const v8h hi8) {
  return __builtin_shufflevector(lo, hi8, 0, 1, 2, 3, 4, 5, 6, 7,
                                 8, 9, 10, 11, 12, 13, 14, 15);
}

// ---- DPP 16-lane butterfly reductions (stay inside each 16-lane half) ----
// xor1 = quad_perm(1,0,3,2)=0xB1, xor2 = quad_perm(2,3,0,1)=0x4E,
// xor7 = row_half_mirror=0x141, xor15 = row_mirror=0x140.
// {1,2,7,15} generates the 16-lane group; max/sum are commutative, so this is
// a full all-lanes reduction with zero LDS traffic (pure VALU DPP).
template <int CTRL>
static __device__ __forceinline__ float dpp16f(float x) {
  return __int_as_float(__builtin_amdgcn_update_dpp(
      0, __float_as_int(x), CTRL, 0xF, 0xF, true));
}
static __device__ __forceinline__ float rmax16(float x) {
  x = fmaxf(x, dpp16f<0xB1>(x));
  x = fmaxf(x, dpp16f<0x4E>(x));
  x = fmaxf(x, dpp16f<0x141>(x));
  x = fmaxf(x, dpp16f<0x140>(x));
  return x;
}
static __device__ __forceinline__ float rsum16(float x) {
  x += dpp16f<0xB1>(x);
  x += dpp16f<0x4E>(x);
  x += dpp16f<0x141>(x);
  x += dpp16f<0x140>(x);
  return x;
}

__global__ __launch_bounds__(128)
void probattn_flash_wmma(const float* __restrict__ Qg,
                         const float* __restrict__ Kg,
                         const float* __restrict__ Vg,
                         float* __restrict__ Og)
{
  __shared__ __align__(16) _Float16 Ksh[BC][KP];      // K chunk, row-major [k][d]
  __shared__ __align__(16) _Float16 Vt[kD][VP];       // V chunk, transposed [d][k]
  __shared__ __align__(16) _Float16 Psh[4][16][PP];   // per-wave P scratch [m][k]

  const int tid  = threadIdx.x;
  const int wave = tid >> 5;
  const int lane = tid & 31;
  const int l16  = lane & 15;
  const int hi   = lane >> 4;          // lane half: 0 or 1

  constexpr int NQT = kL / BR;         // 32 query tiles per (b,h)
  const int bid = blockIdx.x;
  const int qt  = bid % NQT;
  const int h   = (bid / NQT) % kH;
  const int b   = bid / (NQT * kH);

  const int q0 = qt * BR + wave * 16;              // this wave's first q row
  const size_t rowStride = (size_t)kH * kD;        // 512 floats between seq rows

  // ---- Q A-fragments (A layout: lane=M, interleaved K), d=0..31 / 32..63 ----
  v16h qa[2];
  {
    const float* qp = Qg + ((size_t)b * kL + (q0 + l16)) * rowStride + (size_t)h * kD;
#pragma unroll
    for (int f = 0; f < 2; ++f) {
      const int d0 = f * 32 + hi * 8;
#pragma unroll
      for (int i = 0; i < 8; ++i) {
        qa[f][i]     = (_Float16)qp[d0 + i];        // K = d0..d0+7
        qa[f][i + 8] = (_Float16)qp[d0 + 16 + i];   // K = d0+16..d0+23
      }
    }
  }

  // online softmax state: register j holds row (j + 8*hi) of this wave's tile
  float mrow[8], lrow[8];
  v8f o[4];
  const v8f vzero = {};
#pragma unroll
  for (int j = 0; j < 8; ++j) { mrow[j] = -3.0e38f; lrow[j] = 0.f; }
#pragma unroll
  for (int t = 0; t < 4; ++t) o[t] = vzero;

  const float* kb_g = Kg + (size_t)b * kL * rowStride + (size_t)h * kD;
  const float* vb_g = Vg + (size_t)b * kL * rowStride + (size_t)h * kD;

  for (int kc = 0; kc < kL; kc += BC) {
    __syncthreads();   // previous iteration done reading Ksh/Vt
    // ---- cooperative stage: 64 keys. K row-major; V transposed (half2 on k).
    //      512 tasks = 32 row-pairs x 16 float4 columns; 4 tasks/thread. ----
#pragma unroll
    for (int i = 0; i < 4; ++i) {
      const int task = tid + i * 128;    // 0..511
      const int rp   = task >> 4;        // key row pair 0..31
      const int c4   = (task & 15) << 2; // d column 0..60 step 4
      const float* kr0 = kb_g + (size_t)(kc + 2 * rp) * rowStride + c4;
      const float* vr0 = vb_g + (size_t)(kc + 2 * rp) * rowStride + c4;
      const float4 k0 = *(const float4*)(kr0);
      const float4 k1 = *(const float4*)(kr0 + rowStride);
      const float4 v0 = *(const float4*)(vr0);
      const float4 v1 = *(const float4*)(vr0 + rowStride);
      h4 kh0 = { (_Float16)k0.x, (_Float16)k0.y, (_Float16)k0.z, (_Float16)k0.w };
      h4 kh1 = { (_Float16)k1.x, (_Float16)k1.y, (_Float16)k1.z, (_Float16)k1.w };
      *(h4*)&Ksh[2 * rp][c4]     = kh0;
      *(h4*)&Ksh[2 * rp + 1][c4] = kh1;
      h2 vp0 = { (_Float16)v0.x, (_Float16)v1.x };
      h2 vp1 = { (_Float16)v0.y, (_Float16)v1.y };
      h2 vp2 = { (_Float16)v0.z, (_Float16)v1.z };
      h2 vp3 = { (_Float16)v0.w, (_Float16)v1.w };
      *(h2*)&Vt[c4 + 0][2 * rp] = vp0;
      *(h2*)&Vt[c4 + 1][2 * rp] = vp1;
      *(h2*)&Vt[c4 + 2][2 * rp] = vp2;
      *(h2*)&Vt[c4 + 3][2 * rp] = vp3;
      // one global_prefetch_b8 per 128B line of the next chunk
      if ((kc + BC < kL) && ((task & 7) == 0)) {
        __builtin_prefetch(kr0 + (size_t)BC * rowStride, 0, 1);
        __builtin_prefetch(kr0 + (size_t)(BC + 1) * rowStride, 0, 1);
        __builtin_prefetch(vr0 + (size_t)BC * rowStride, 0, 1);
        __builtin_prefetch(vr0 + (size_t)(BC + 1) * rowStride, 0, 1);
      }
    }
    __syncthreads();

    // ---- S = Q K^T : four 16x16 key subtiles, K-dim split 2x32 over d ----
    v8f s[4];
#pragma unroll
    for (int n = 0; n < 4; ++n) {
      const _Float16* kr = &Ksh[n * 16 + l16][0];
      // B layout: lane = column (key), half idx -> K(d) interleaved: 8*hi + (0..7), 16 + 8*hi + (0..7)
      v16h kb0 = cat8(*(const v8h*)&kr[hi * 8],      *(const v8h*)&kr[hi * 8 + 16]);
      v16h kb1 = cat8(*(const v8h*)&kr[32 + hi * 8], *(const v8h*)&kr[32 + hi * 8 + 16]);
      v8f acc = vzero;
      acc = __builtin_amdgcn_wmma_f32_16x16x32_f16(false, qa[0], false, kb0, 0, acc, false, false);
      acc = __builtin_amdgcn_wmma_f32_16x16x32_f16(false, qa[1], false, kb1, 0, acc, false, false);
      s[n] = acc;
    }

    // ---- online softmax; DPP butterfly reductions, no LDS traffic ----
#pragma unroll
    for (int j = 0; j < 8; ++j) {
      float vmax = fmaxf(fmaxf(s[0][j], s[1][j]), fmaxf(s[2][j], s[3][j]));
      vmax = rmax16(vmax);
      const float mnew = fmaxf(mrow[j], vmax);
      const float corr = __expf(mrow[j] - mnew);
      float p[4];
      float rs = 0.f;
#pragma unroll
      for (int n = 0; n < 4; ++n) {
        p[n] = __expf(s[n][j] - mnew);   // NOTE: no 1/sqrt(D), per reference
        rs += p[n];
      }
      rs = rsum16(rs);
      lrow[j] = lrow[j] * corr + rs;
      mrow[j] = mnew;
#pragma unroll
      for (int t = 0; t < 4; ++t) o[t][j] *= corr;   // rescale O accumulator
#pragma unroll
      for (int n = 0; n < 4; ++n)                     // C-layout -> LDS
        Psh[wave][hi * 8 + j][n * 16 + l16] = (_Float16)p[n];
    }

    // ---- reload P as two A-fragments (wave-local LDS, in-order) ----
    const _Float16* pr = &Psh[wave][l16][0];
    v16h pa0 = cat8(*(const v8h*)&pr[hi * 8],      *(const v8h*)&pr[hi * 8 + 16]);
    v16h pa1 = cat8(*(const v8h*)&pr[32 + hi * 8], *(const v8h*)&pr[32 + hi * 8 + 16]);

    // ---- O += P * V : two WMMAs per 16-wide d tile (K = 64 keys) ----
#pragma unroll
    for (int t = 0; t < 4; ++t) {
      const _Float16* vr = &Vt[t * 16 + l16][0];   // lane = output column d
      v16h vb0 = cat8(*(const v8h*)&vr[hi * 8],      *(const v8h*)&vr[hi * 8 + 16]);
      v16h vb1 = cat8(*(const v8h*)&vr[32 + hi * 8], *(const v8h*)&vr[32 + hi * 8 + 16]);
      o[t] = __builtin_amdgcn_wmma_f32_16x16x32_f16(false, pa0, false, vb0, 0, o[t], false, false);
      o[t] = __builtin_amdgcn_wmma_f32_16x16x32_f16(false, pa1, false, vb1, 0, o[t], false, false);
    }
  }

  // ---- normalize and store to (B, L, H, D) ----
#pragma unroll
  for (int j = 0; j < 8; ++j) {
    const float inv = 1.0f / lrow[j];
    float* orow = Og + ((size_t)b * kL + (q0 + hi * 8 + j)) * rowStride + (size_t)h * kD;
#pragma unroll
    for (int t = 0; t < 4; ++t)
      orow[t * 16 + l16] = o[t][j] * inv;
  }
}

extern "C" void kernel_launch(void* const* d_in, const int* in_sizes, int n_in,
                              void* d_out, int out_size, void* d_ws, size_t ws_size,
                              hipStream_t stream) {
  const float* Q = (const float*)d_in[0];
  const float* K = (const float*)d_in[1];
  const float* V = (const float*)d_in[2];
  float* O = (float*)d_out;
  const int grid = kB * kH * (kL / BR);   // 1024 workgroups, 128 threads (4 waves)
  probattn_flash_wmma<<<grid, 128, 0, stream>>>(Q, K, V, O);
}
